// CausalMultiHeadAttentionLayer_9569187136112
// MI455X (gfx1250) — matrix operand order repaired and checked by reading, not compiled, not run on hardware
//
#include <hip/hip_runtime.h>
#include <hip/hip_fp16.h>

// ---------------------------------------------------------------------------
// Causal MHA for MI455X (gfx1250): f16 WMMA GEMMs + flash attention.
// B=4, S=2048, D=1024, H=16, DH=64
// GEMMs: 256x128 block tile, 64x64 wave tile, double-buffered LDS with
// async global->LDS staging (ASYNCcnt). Attention: 128 q-rows/block (8 waves),
// 64-key blocks, online softmax; all matrix math on v_wmma_f32_16x16x32_f16.
// B-fragment loads are batched ahead of each WMMA group so the scheduler can
// cover LDS latency with one s_wait_dscnt per group instead of one per WMMA.
// ---------------------------------------------------------------------------

typedef __attribute__((ext_vector_type(16))) _Float16 v16h;
typedef __attribute__((ext_vector_type(8)))  _Float16 v8h;
typedef __attribute__((ext_vector_type(8)))  float    v8f;

constexpr int B_  = 4;
constexpr int S_  = 2048;
constexpr int D_  = 1024;
constexpr int H_  = 16;
constexpr int DH_ = 64;
constexpr int M_  = B_ * S_;          // 8192

// workspace layout (bytes, all 256-aligned)
constexpr size_t XH_OFF   = 0;                              // x f16 [8192][1024]
constexpr size_t WQKV_OFF = XH_OFF   + ((size_t)16 << 20);  // Wq|Wk|Wv f16 [3072][1024]
constexpr size_t WO_OFF   = WQKV_OFF + ((size_t)6  << 20);  // Wo f16 [1024][1024]
constexpr size_t QH_OFF   = WO_OFF   + ((size_t)2  << 20);  // Q f16 [B,H,S,DH]
constexpr size_t KH_OFF   = QH_OFF   + ((size_t)16 << 20);  // K f16
constexpr size_t VH_OFF   = KH_OFF   + ((size_t)16 << 20);  // V f16
constexpr size_t OH_OFF   = VH_OFF   + ((size_t)16 << 20);  // attn out f16 [8192][1024]

// ---------------------------------------------------------------------------
// CDNA5 async global->LDS copy (ASYNCcnt) with synchronous fallback
// ---------------------------------------------------------------------------
#if __has_builtin(__builtin_amdgcn_global_load_async_to_lds_b128)
#define HAS_ASYNC_LDS 1
#else
#define HAS_ASYNC_LDS 0
#endif

typedef int v4i __attribute__((vector_size(16)));
typedef __attribute__((address_space(1))) v4i* gv4i_p;   // global int4*
typedef __attribute__((address_space(3))) v4i* lv4i_p;   // LDS    int4*

__device__ __forceinline__ void async_copy_b128(const _Float16* g, _Float16* l) {
#if HAS_ASYNC_LDS
  __builtin_amdgcn_global_load_async_to_lds_b128(
      (gv4i_p)(uintptr_t)g, (lv4i_p)(uint32_t)(uintptr_t)l, 0, 0);
#else
  *(uint4*)l = *(const uint4*)g;
#endif
}

__device__ __forceinline__ void async_wait_all() {
#if HAS_ASYNC_LDS
#if __has_builtin(__builtin_amdgcn_s_wait_asynccnt)
  __builtin_amdgcn_s_wait_asynccnt(0);
#else
  asm volatile("s_wait_asynccnt 0" ::: "memory");
#endif
#endif
}

// ---------------------------------------------------------------------------
// WMMA helpers (layouts per CDNA5 ISA 7.12.2)
// ---------------------------------------------------------------------------
__device__ __forceinline__ v8f wmma_f16(v16h a, v16h b, v8f c) {
  return __builtin_amdgcn_wmma_f32_16x16x32_f16(
      false, a, false, b, (short)0, c, false, false);
}

// A fragment 16x32 f16 from row-major [rows][ld] halfs.
// lane<16: K {0..7,16..23}; lane>=16: K {8..15,24..31}; row = lane%16.
__device__ __forceinline__ v16h load_frag_a(const _Float16* base, int ld, int lane) {
  const _Float16* p = base + (lane & 15) * ld + ((lane < 16) ? 0 : 8);
  v8h lo = *(const v8h*)(p);
  v8h hi = *(const v8h*)(p + 16);
  return __builtin_shufflevector(lo, hi, 0,1,2,3,4,5,6,7,8,9,10,11,12,13,14,15);
}

// B fragment 32x16 f16 where column n is contiguous in K: B[k][n] at base[n*ld+k].
// lanes 0-15: K {0..7,16..23}; lanes 16-31: K {8..15,24..31}; n = lane%16.
__device__ __forceinline__ v16h load_frag_b(const _Float16* base, int ld, int lane) {
  const _Float16* p = base + (lane & 15) * ld + ((lane < 16) ? 0 : 8);
  v8h lo = *(const v8h*)(p);
  v8h hi = *(const v8h*)(p + 16);
  return __builtin_shufflevector(lo, hi, 0,1,2,3,4,5,6,7,8,9,10,11,12,13,14,15);
}

__device__ __forceinline__ float hred_max(float v) {
  v = fmaxf(v, __shfl_xor(v, 1, 32));
  v = fmaxf(v, __shfl_xor(v, 2, 32));
  v = fmaxf(v, __shfl_xor(v, 4, 32));
  v = fmaxf(v, __shfl_xor(v, 8, 32));
  return v;
}
__device__ __forceinline__ float hred_sum(float v) {
  v += __shfl_xor(v, 1, 32);
  v += __shfl_xor(v, 2, 32);
  v += __shfl_xor(v, 4, 32);
  v += __shfl_xor(v, 8, 32);
  return v;
}

// ---------------------------------------------------------------------------
// conversion kernels (8 elements / thread, b128 stores)
// ---------------------------------------------------------------------------
__global__ void cvt_f32_to_f16_x8(const float* __restrict__ in, _Float16* __restrict__ out) {
  int i = blockIdx.x * blockDim.x + threadIdx.x;      // one v8h per thread
  float4 f0 = *(const float4*)(in + i * 8);
  float4 f1 = *(const float4*)(in + i * 8 + 4);
  v8h o = {(_Float16)f0.x, (_Float16)f0.y, (_Float16)f0.z, (_Float16)f0.w,
           (_Float16)f1.x, (_Float16)f1.y, (_Float16)f1.z, (_Float16)f1.w};
  *((v8h*)out + i) = o;
}

__global__ void pack_wqkv_x8(const float* __restrict__ wq, const float* __restrict__ wk,
                             const float* __restrict__ wv, _Float16* __restrict__ out) {
  int i = blockIdx.x * blockDim.x + threadIdx.x;      // 0 .. 3*1024*1024/8 - 1
  int which = i >> 17;
  int rem8  = (i & 0x1FFFF) << 3;
  const float* s = (which == 0) ? wq : (which == 1) ? wk : wv;
  float4 f0 = *(const float4*)(s + rem8);
  float4 f1 = *(const float4*)(s + rem8 + 4);
  v8h o = {(_Float16)f0.x, (_Float16)f0.y, (_Float16)f0.z, (_Float16)f0.w,
           (_Float16)f1.x, (_Float16)f1.y, (_Float16)f1.z, (_Float16)f1.w};
  *((v8h*)out + i) = o;
}

// ---------------------------------------------------------------------------
// WMMA GEMM: Y[m][n] = sum_k A[m][k] * W[n][k]   (y = x @ W^T)
// Block 256 thr (8 waves), block tile 256x128, wave tile 64x64, k-step 32.
// ---------------------------------------------------------------------------
__device__ __forceinline__ void stage_gemm(const _Float16* __restrict__ A,
                                           const _Float16* __restrict__ W,
                                           _Float16* as, _Float16* bs,
                                           int mblk, int nblk, int k0, int tid) {
#pragma unroll
  for (int i = 0; i < 4; ++i) {                 // A tile 256x32 = 1024 b128 units
    int u = tid + i * 256;
    int row = u >> 2, co = (u & 3) * 8;
    async_copy_b128(A + (size_t)(mblk + row) * 1024 + k0 + co, as + row * 32 + co);
  }
#pragma unroll
  for (int i = 0; i < 2; ++i) {                 // W tile 128x32 = 512 b128 units
    int u = tid + i * 256;
    int row = u >> 2, co = (u & 3) * 8;
    async_copy_b128(W + (size_t)(nblk + row) * 1024 + k0 + co, bs + row * 32 + co);
  }
}

template <int MODE>
__global__ __launch_bounds__(256)
void gemm_xwT(const _Float16* __restrict__ A, const _Float16* __restrict__ W,
              _Float16* __restrict__ q, _Float16* __restrict__ k, _Float16* __restrict__ v,
              const float* __restrict__ bias, float* __restrict__ outf) {
  __shared__ alignas(16) _Float16 As[2][256 * 32];    // 2 x 16 KiB
  __shared__ alignas(16) _Float16 Bs[2][128 * 32];    // 2 x  8 KiB

  const int tid  = threadIdx.x;
  const int lane = tid & 31;
  const int wid  = tid >> 5;
  const int wm   = wid >> 1;                 // 0..3 -> 64-row slab
  const int wn   = wid & 1;                  // 0..1 -> 64-col slab
  const int mblk = blockIdx.y * 256;
  const int nblk = blockIdx.x * 128;

  v8f acc[4][4] = {};

  stage_gemm(A, W, As[0], Bs[0], mblk, nblk, 0, tid);
  async_wait_all();
  __syncthreads();

  for (int k0 = 0; k0 < 1024; k0 += 32) {
    const int cur = (k0 >> 5) & 1;
    if (k0 + 32 < 1024)
      stage_gemm(A, W, As[cur ^ 1], Bs[cur ^ 1], mblk, nblk, k0 + 32, tid);

    // batch all fragment loads first, then the 16-WMMA burst
    v16h a[4], bf[4];
#pragma unroll
    for (int tm = 0; tm < 4; ++tm)
      a[tm] = load_frag_a(&As[cur][(wm * 64 + tm * 16) * 32], 32, lane);
#pragma unroll
    for (int tn = 0; tn < 4; ++tn)
      bf[tn] = load_frag_b(&Bs[cur][(wn * 64 + tn * 16) * 32], 32, lane);
#pragma unroll
    for (int tn = 0; tn < 4; ++tn)
#pragma unroll
      for (int tm = 0; tm < 4; ++tm)
        acc[tm][tn] = wmma_f16(a[tm], bf[tn], acc[tm][tn]);

    async_wait_all();
    __syncthreads();
  }

  // epilogue: C layout -> VGPR r, lane n: row = r + 8*(n>=16), col = n%16
  const int roff = (lane >> 4) << 3;
  const int cl   = lane & 15;
#pragma unroll
  for (int tm = 0; tm < 4; ++tm) {
#pragma unroll
    for (int tn = 0; tn < 4; ++tn) {
#pragma unroll
      for (int r = 0; r < 8; ++r) {
        int row = mblk + wm * 64 + tm * 16 + r + roff;
        int ng  = nblk + wn * 64 + tn * 16 + cl;
        float val = acc[tm][tn][r];
        if (MODE == 0) {
          int which = ng >> 10;              // 0:Q 1:K 2:V
          int nn    = ng & 1023;
          int h     = nn >> 6;
          int dh    = nn & 63;
          int b     = row >> 11;
          int s     = row & 2047;
          _Float16* dst = (which == 0) ? q : (which == 1) ? k : v;
          dst[(((size_t)(b * H_ + h)) * S_ + s) * DH_ + dh] = (_Float16)val;
        } else {
          outf[(size_t)row * 1024 + ng] = val + bias[ng];
        }
      }
    }
  }
}

// ---------------------------------------------------------------------------
// Flash attention: one (b,h) per blockIdx.y, 128 query rows per block,
// 8 waves x 16 q-rows, 64-key blocks. kend = blk*128+128 is block-uniform so every wave is
// active every iteration (uniform control flow, EXEC all-1s for WMMA).
// ---------------------------------------------------------------------------
__global__ __launch_bounds__(256)
void flash_attn(const _Float16* __restrict__ qh, const _Float16* __restrict__ kh,
                const _Float16* __restrict__ vh, _Float16* __restrict__ oh) {
  __shared__ alignas(16) _Float16 kt[64 * 64];       // K  [key][dh]       8 KiB
  __shared__ alignas(16) _Float16 vt[64 * 64];       // V^T [dh][key]      8 KiB
  __shared__ alignas(16) _Float16 ps[8][16 * 64];    // per-wave P        16 KiB

  const int tid  = threadIdx.x;
  const int lane = tid & 31;
  const int wid  = tid >> 5;
  const int bh   = blockIdx.y;
  const int b    = bh >> 4;
  const int h    = bh & 15;
  const int q0   = blockIdx.x * 128 + wid * 16;

  const _Float16* Qb = qh + (size_t)bh * S_ * DH_;
  const _Float16* Kb = kh + (size_t)bh * S_ * DH_;
  const _Float16* Vb = vh + (size_t)bh * S_ * DH_;

  // Q tile resident in registers for the whole key loop
  v16h qf0 = load_frag_a(Qb + (size_t)q0 * DH_,      DH_, lane);   // dh 0..31
  v16h qf1 = load_frag_a(Qb + (size_t)q0 * DH_ + 32, DH_, lane);   // dh 32..63

  v8f   o[4] = {};
  float mi[8], li[8];
#pragma unroll
  for (int r = 0; r < 8; ++r) { mi[r] = -__builtin_inff(); li[r] = 0.0f; }

  const int roff = (lane >> 4) << 3;
  const int cl   = lane & 15;
  const int kend = blockIdx.x * 128 + 128;

  for (int kb = 0; kb < kend; kb += 64) {
    // ---- stage K tile (async b128) and V^T tile (coalesced read + scatter)
#pragma unroll
    for (int i = 0; i < 2; ++i) {              // 64x64 halfs = 512 b128 units
      int u = tid + i * 256;
      int row = u >> 3, co = (u & 7) * 8;
      async_copy_b128(Kb + (size_t)(kb + row) * DH_ + co, &kt[row * 64 + co]);
    }
    {
      int key = tid >> 2, ch = (tid & 3) * 16;
      const v8h* src = (const v8h*)(Vb + (size_t)(kb + key) * DH_ + ch);
      v8h t0 = src[0], t1 = src[1];
#pragma unroll
      for (int j = 0; j < 8; ++j) {
        vt[(ch + j)     * 64 + key] = t0[j];
        vt[(ch + j + 8) * 64 + key] = t1[j];
      }
    }
    async_wait_all();
    __syncthreads();

    // ---- S = Q K^T : four 16x16 score tiles, DH=64 = 2 wmma k-steps ------
    v8f  s[4];
    v16h bk[4];
#pragma unroll
    for (int ti = 0; ti < 4; ++ti)
      bk[ti] = load_frag_b(&kt[(ti * 16) * 64], 64, lane);
#pragma unroll
    for (int ti = 0; ti < 4; ++ti) { v8f z = {}; s[ti] = wmma_f16(qf0, bk[ti], z); }
#pragma unroll
    for (int ti = 0; ti < 4; ++ti)
      bk[ti] = load_frag_b(&kt[(ti * 16) * 64 + 32], 64, lane);
#pragma unroll
    for (int ti = 0; ti < 4; ++ti) s[ti] = wmma_f16(qf1, bk[ti], s[ti]);

    // ---- masked online softmax ------------------------------------------
    const float sc = 1.0f / 64.0f;             // reference scales by 1/DH
#pragma unroll
    for (int r = 0; r < 8; ++r) {
      int row = q0 + r + roff;
      float a[4];
#pragma unroll
      for (int ti = 0; ti < 4; ++ti) {
        int c = kb + ti * 16 + cl;
        a[ti] = s[ti][r] * sc;
        if (c > row) a[ti] = -__builtin_inff();
      }
      float mx = hred_max(fmaxf(fmaxf(a[0], a[1]), fmaxf(a[2], a[3])));
      float mn = fmaxf(mi[r], mx);
      float p[4];
#pragma unroll
      for (int ti = 0; ti < 4; ++ti) p[ti] = __expf(a[ti] - mn);
      float rs = hred_sum((p[0] + p[1]) + (p[2] + p[3]));
      float al = __expf(mi[r] - mn);
      li[r] = li[r] * al + rs;
      mi[r] = mn;
#pragma unroll
      for (int t = 0; t < 4; ++t) o[t][r] *= al;
      int prow = r + roff;
#pragma unroll
      for (int ti = 0; ti < 4; ++ti)
        ps[wid][prow * 64 + ti * 16 + cl] = (_Float16)p[ti];
    }

    // re-layout P (C-layout -> A-fragment) through per-wave LDS scratch
    asm volatile("s_wait_dscnt 0" ::: "memory");
    v16h pa0 = load_frag_a(&ps[wid][0],  64, lane);   // keys kb+0..31
    v16h pa1 = load_frag_a(&ps[wid][32], 64, lane);   // keys kb+32..63

    // ---- O += P V : 64-key k-dim = 2 wmma steps per 16-col tile ----------
    v16h bv[4];
#pragma unroll
    for (int t = 0; t < 4; ++t)
      bv[t] = load_frag_b(&vt[(t * 16) * 64], 64, lane);
#pragma unroll
    for (int t = 0; t < 4; ++t) o[t] = wmma_f16(pa0, bv[t], o[t]);
#pragma unroll
    for (int t = 0; t < 4; ++t)
      bv[t] = load_frag_b(&vt[(t * 16) * 64 + 32], 64, lane);
#pragma unroll
    for (int t = 0; t < 4; ++t) o[t] = wmma_f16(pa1, bv[t], o[t]);

    __syncthreads();
  }

  // ---- normalize and write attn output as f16 [B*S][D] -------------------
#pragma unroll
  for (int r = 0; r < 8; ++r) {
    float inv = 1.0f / li[r];
    int s = q0 + r + roff;
#pragma unroll
    for (int t = 0; t < 4; ++t) {
      int col = h * DH_ + t * 16 + cl;
      oh[((size_t)(b * S_ + s)) * D_ + col] = (_Float16)(o[t][r] * inv);
    }
  }
}

// ---------------------------------------------------------------------------
extern "C" void kernel_launch(void* const* d_in, const int* in_sizes, int n_in,
                              void* d_out, int out_size, void* d_ws, size_t ws_size,
                              hipStream_t stream) {
  const float* x  = (const float*)d_in[0];
  const float* Wq = (const float*)d_in[1];
  const float* Wk = (const float*)d_in[2];
  const float* Wv = (const float*)d_in[3];
  const float* Wo = (const float*)d_in[4];
  const float* bo = (const float*)d_in[5];
  float* out = (float*)d_out;

  char* ws = (char*)d_ws;
  _Float16* xh   = (_Float16*)(ws + XH_OFF);
  _Float16* wqkv = (_Float16*)(ws + WQKV_OFF);
  _Float16* woh  = (_Float16*)(ws + WO_OFF);
  _Float16* qhp  = (_Float16*)(ws + QH_OFF);
  _Float16* khp  = (_Float16*)(ws + KH_OFF);
  _Float16* vhp  = (_Float16*)(ws + VH_OFF);
  _Float16* ohp  = (_Float16*)(ws + OH_OFF);

  // 1) precision conversion / packing
  cvt_f32_to_f16_x8<<<(M_ * D_) / 8 / 256, 256, 0, stream>>>(x, xh);
  pack_wqkv_x8<<<(3 * D_ * D_) / 8 / 256, 256, 0, stream>>>(Wq, Wk, Wv, wqkv);
  cvt_f32_to_f16_x8<<<(D_ * D_) / 8 / 256, 256, 0, stream>>>(Wo, woh);

  // 2) fused Q/K/V projection GEMM: [8192x1024] x [3072x1024]^T
  gemm_xwT<0><<<dim3(3072 / 128, M_ / 256), 256, 0, stream>>>(
      xh, wqkv, qhp, khp, vhp, nullptr, nullptr);

  // 3) causal flash attention per (b,h)
  flash_attn<<<dim3(S_ / 128, B_ * H_), 256, 0, stream>>>(qhp, khp, vhp, ohp);

  // 4) output projection + bias, fp32 result
  gemm_xwT<1><<<dim3(1024 / 128, M_ / 256), 256, 0, stream>>>(
      ohp, woh, nullptr, nullptr, nullptr, bo, out);

  (void)in_sizes; (void)n_in; (void)out_size; (void)ws_size;
}
